// Qwen3VLTextAttention_80470507258415
// MI455X (gfx1250) — matrix operand order repaired and checked by reading, not compiled, not run on hardware
//
#include <hip/hip_runtime.h>

#define DEV __device__ __forceinline__

typedef __attribute__((ext_vector_type(16))) __bf16 v16bf;
typedef __attribute__((ext_vector_type(8)))  __bf16 v8bf;
typedef __attribute__((ext_vector_type(8)))  float  v8f;
typedef __attribute__((ext_vector_type(4)))  unsigned int u32x4;
typedef __attribute__((ext_vector_type(8)))  int i32x8;
typedef __attribute__((ext_vector_type(4)))  int i32x4;

static constexpr int S_LEN = 2048;
static constexpr int DIM   = 4096;
static constexpr int NH    = 32;
static constexpr int NKV   = 8;
static constexpr int HD    = 128;

DEV __bf16 f2bf(float f) {
  union { float f; unsigned u; } uf; uf.f = f;
  unsigned u = uf.u;
  u += 0x7FFFu + ((u >> 16) & 1u);            // round-to-nearest-even
  unsigned short h = (unsigned short)(u >> 16);
  union { unsigned short s; __bf16 b; } ub; ub.s = h;
  return ub.b;
}

// A-fragment load, contiguous k order (matches B loads):
// lane(0-15): row M=lane, elems 0..15 -> K = hi*16+0..15
DEV v16bf load_a(const __bf16* base, int hi) {
  v8bf lo = *(const v8bf*)(base + hi * 16);
  v8bf hh = *(const v8bf*)(base + hi * 16 + 8);
  return __builtin_shufflevector(lo, hh, 0,1,2,3,4,5,6,7,8,9,10,11,12,13,14,15);
}

DEV v8f wmma_bf16(v16bf a, v16bf b, v8f c) {
  return __builtin_amdgcn_wmma_f32_16x16x32_bf16(false, a, false, b, (short)0, c, false, false);
}

// ---------------------------------------------------------------------------
// Tensor Data Mover: DMA a 64(rows) x 64(cols) bf16 tile of W[N,K] into LDS.
// D# per CDNA5 ISA ch.8: group0 {count, lds_addr, global_addr, type=2},
// group1 {data_size=2B, tensor dims/strides, tile dims}, groups 2/3 unused (2D).
// ---------------------------------------------------------------------------
DEV void tdm_load_tile_b(const __bf16* W, int N, int K, int n0, int k0,
                         unsigned int lds_byte_addr) {
  unsigned long long gaddr = (unsigned long long)(uintptr_t)(W + (size_t)n0 * K + k0);
  u32x4 g0;
  g0[0] = 1u;                                            // count=1, user descriptor
  g0[1] = lds_byte_addr;                                 // lds_addr[31:0]
  g0[2] = (unsigned int)gaddr;                           // global_addr[31:0]
  g0[3] = (unsigned int)((gaddr >> 32) & 0x01FFFFFFu)    // global_addr[56:32]
          | (2u << 30);                                  // type=2 ("image")
  i32x8 g1;
  g1[0] = 0x00010000;                                    // wg_mask=0, data_size=1 (2 bytes)
  g1[1] = (int)((K & 0xFFFF) << 16);                     // tensor_dim0[15:0]
  g1[2] = (int)(((unsigned)K >> 16) | ((unsigned)(N & 0xFFFF) << 16)); // dim0 hi | dim1 lo
  g1[3] = (int)(((unsigned)N >> 16) | (64u << 16));      // dim1 hi | tile_dim0=64
  g1[4] = 64;                                            // tile_dim1=64, tile_dim2=0
  g1[5] = (int)K;                                        // tensor_dim0_stride[31:0]
  g1[6] = 0;                                             // stride hi / dim1_stride lo
  g1[7] = 0;
  i32x4 z4 = (i32x4)0;
#if __clang_major__ >= 23
  i32x8 z8 = (i32x8)0;
  __builtin_amdgcn_tensor_load_to_lds(g0, g1, z4, z4, z8, 0);
#else
  __builtin_amdgcn_tensor_load_to_lds(g0, g1, z4, z4, 0);
#endif
}

// ---------------------------------------------------------------------------
// fp32 -> bf16 cast
// ---------------------------------------------------------------------------
__global__ void cvt_bf16_kernel(const float* __restrict__ in, __bf16* __restrict__ out, long n) {
  long i = (long)blockIdx.x * blockDim.x + threadIdx.x;
  if (i < n) out[i] = f2bf(in[i]);
}

// ---------------------------------------------------------------------------
// GEMM: Y[M,N] = X[M,K] * W[N,K]^T   (bf16 in, fp32/bf16 out)
// Block = 8 waves = 128(M) x 64(N) tile. B tile (64x64) staged into LDS by the
// Tensor Data Mover, double-buffered, synchronized with s_wait_tensorcnt.
// grid = (N/64, M/128)
// ---------------------------------------------------------------------------
template <bool BF16_OUT>
__global__ __launch_bounds__(256) void gemm_xwT_tdm(const __bf16* __restrict__ X,
                                                    const __bf16* __restrict__ W,
                                                    void* __restrict__ Yv,
                                                    int M, int N, int K) {
  __shared__ __bf16 Bs[2][64 * 64];                      // 2 x 8 KB
  const int lane = threadIdx.x & 31;
  const int wv   = threadIdx.x >> 5;
  const int lo16 = lane & 15;
  const int hi   = lane >> 4;
  const int m0   = blockIdx.y * 128 + wv * 16;
  const int n0   = blockIdx.x * 64;

  v8f acc[4];
  #pragma unroll
  for (int j = 0; j < 4; ++j) acc[j] = (v8f)0.f;

  const __bf16* xrow = X + (size_t)(m0 + lo16) * K;
  const unsigned int lds0 = (unsigned int)(uintptr_t)&Bs[0][0];
  const unsigned int lds1 = (unsigned int)(uintptr_t)&Bs[1][0];

  const int KT = K >> 6;                                 // K / 64 steps
  if (threadIdx.x == 0) tdm_load_tile_b(W, N, K, n0, 0, lds0);

  for (int kt = 0; kt < KT; ++kt) {
    const int cur = kt & 1;
    if (threadIdx.x == 0) {
      if (kt + 1 < KT) {
        tdm_load_tile_b(W, N, K, n0, (kt + 1) << 6, cur ? lds0 : lds1);
        __builtin_amdgcn_s_wait_tensorcnt(1);            // current tile complete
      } else {
        __builtin_amdgcn_s_wait_tensorcnt(0);            // last tile complete
      }
    }
    __syncthreads();                                     // publish B tile

    const __bf16* bs = &Bs[cur][0];
    #pragma unroll
    for (int ks = 0; ks < 2; ++ks) {
      v16bf a = load_a(xrow + (kt << 6) + ks * 32, hi);
      #pragma unroll
      for (int j = 0; j < 4; ++j) {
        v16bf b = *(const v16bf*)(bs + (j * 16 + lo16) * 64 + ks * 32 + hi * 16);
        acc[j] = wmma_bf16(a, b, acc[j]);
      }
    }
    __syncthreads();                                     // done reading before overwrite
  }

  // C layout: VGPR r -> row m0 + hi*8 + r, col n = lane&15
  #pragma unroll
  for (int j = 0; j < 4; ++j) {
    #pragma unroll
    for (int r = 0; r < 8; ++r) {
      size_t idx = (size_t)(m0 + hi * 8 + r) * N + n0 + j * 16 + lo16;
      if (BF16_OUT) ((__bf16*)Yv)[idx] = f2bf(acc[j][r]);
      else          ((float*)Yv)[idx]  = acc[j][r];
    }
  }
}

// ---------------------------------------------------------------------------
// RMSNorm + RoPE, fused. One wave per (s, head). X fp32 [S,H,128] -> Y bf16.
// ---------------------------------------------------------------------------
__global__ __launch_bounds__(32) void rmsnorm_rope_kernel(const float* __restrict__ X,
                                                          const float* __restrict__ cosb,
                                                          const float* __restrict__ sinb,
                                                          const float* __restrict__ w,
                                                          __bf16* __restrict__ Y, int H) {
  const int lane = threadIdx.x;
  const int s = blockIdx.x;
  const int h = blockIdx.y;
  const float* xr = X + ((size_t)s * H + h) * HD;
  float x0 = xr[lane], x1 = xr[lane + 32], x2 = xr[lane + 64], x3 = xr[lane + 96];
  float ss = x0 * x0 + x1 * x1 + x2 * x2 + x3 * x3;
  #pragma unroll
  for (int i = 1; i < 32; i <<= 1) ss += __shfl_xor(ss, i, 32);
  float rinv = rsqrtf(ss * (1.0f / 128.0f) + 1e-6f);
  float n0 = x0 * rinv * w[lane];
  float n1 = x1 * rinv * w[lane + 32];
  float n2 = x2 * rinv * w[lane + 64];
  float n3 = x3 * rinv * w[lane + 96];
  const float* cr = cosb + (size_t)s * HD;
  const float* sr = sinb + (size_t)s * HD;
  __bf16* yr = Y + ((size_t)s * H + h) * HD;
  yr[lane]      = f2bf(n0 * cr[lane]      - n2 * sr[lane]);
  yr[lane + 32] = f2bf(n1 * cr[lane + 32] - n3 * sr[lane + 32]);
  yr[lane + 64] = f2bf(n2 * cr[lane + 64] + n0 * sr[lane + 64]);
  yr[lane + 96] = f2bf(n3 * cr[lane + 96] + n1 * sr[lane + 96]);
}

// ---------------------------------------------------------------------------
// V transpose: [S, NKV, 128] -> [NKV, 128, S] (bf16), output-coalesced
// ---------------------------------------------------------------------------
__global__ void transpose_v_kernel(const __bf16* __restrict__ V, __bf16* __restrict__ Vt) {
  long i = (long)blockIdx.x * blockDim.x + threadIdx.x;  // over NKV*HD*S
  int s = (int)(i & (S_LEN - 1));
  long dh = i >> 11;
  int d = (int)(dh & (HD - 1));
  int kvh = (int)(dh >> 7);
  Vt[i] = V[((size_t)s * NKV + kvh) * HD + d];
}

// ---------------------------------------------------------------------------
// Flash attention: 1 wave per (16-query tile, head). Causal, online softmax.
// Q [S,NH,128] bf16, K [S,NKV,128] bf16, Vt [NKV,128,S] bf16 -> O [S,NH,128] bf16
// ---------------------------------------------------------------------------
__global__ __launch_bounds__(32) void flash_attn_kernel(const __bf16* __restrict__ Q,
                                                        const __bf16* __restrict__ Kd,
                                                        const __bf16* __restrict__ Vt,
                                                        __bf16* __restrict__ O) {
  __shared__ __bf16 Pl[16 * 32];
  const int lane = threadIdx.x & 31;
  const int lo16 = lane & 15;
  const int hi   = lane >> 4;
  const int h    = blockIdx.y;
  const int kvh  = h >> 2;           // n_rep = 4
  const int q0   = blockIdx.x * 16;
  const float scale = 0.08838834764831845f;  // 1/sqrt(128)

  v16bf aq[4];
  {
    const __bf16* qrow = Q + ((size_t)(q0 + lo16) * NH + h) * HD;
    #pragma unroll
    for (int c = 0; c < 4; ++c) aq[c] = load_a(qrow + c * 32, hi);
  }

  v8f o[8];
  #pragma unroll
  for (int j = 0; j < 8; ++j) o[j] = (v8f)0.f;
  float mrow[8], lrow[8];
  #pragma unroll
  for (int r = 0; r < 8; ++r) { mrow[r] = -3.0e38f; lrow[r] = 0.f; }

  for (int kb = 0; kb < q0 + 16; kb += 32) {
    // ---- scores: two 16-key tiles, K=128 contraction ----
    v8f st0 = (v8f)0.f, st1 = (v8f)0.f;
    {
      const __bf16* kr0 = Kd + ((size_t)(kb + lo16)      * NKV + kvh) * HD + hi * 16;
      const __bf16* kr1 = Kd + ((size_t)(kb + 16 + lo16) * NKV + kvh) * HD + hi * 16;
      #pragma unroll
      for (int c = 0; c < 4; ++c) {
        st0 = wmma_bf16(aq[c], *(const v16bf*)(kr0 + c * 32), st0);
        st1 = wmma_bf16(aq[c], *(const v16bf*)(kr1 + c * 32), st1);
      }
    }

    // ---- causal mask + online softmax (per-row stats across 16-lane half) ----
    const int key0 = kb + lo16;
    const int key1 = kb + 16 + lo16;
    #pragma unroll
    for (int r = 0; r < 8; ++r) {
      const int qr = q0 + hi * 8 + r;
      float s0 = (key0 <= qr) ? st0[r] * scale : -3.0e38f;
      float s1 = (key1 <= qr) ? st1[r] * scale : -3.0e38f;
      float mx = fmaxf(s0, s1);
      #pragma unroll
      for (int i = 1; i < 16; i <<= 1) mx = fmaxf(mx, __shfl_xor(mx, i, 16));
      float mnew = fmaxf(mrow[r], mx);
      float corr = __expf(mrow[r] - mnew);
      float p0 = __expf(s0 - mnew);
      float p1 = __expf(s1 - mnew);
      float ps = p0 + p1;
      #pragma unroll
      for (int i = 1; i < 16; i <<= 1) ps += __shfl_xor(ps, i, 16);
      lrow[r] = lrow[r] * corr + ps;
      mrow[r] = mnew;
      #pragma unroll
      for (int j = 0; j < 8; ++j) o[j][r] *= corr;
      Pl[(hi * 8 + r) * 32 + lo16]      = f2bf(p0);
      Pl[(hi * 8 + r) * 32 + 16 + lo16] = f2bf(p1);
    }

    // ---- P * V: reload P as A-fragment from LDS, V^T gives contiguous B ----
    v16bf pa = load_a(&Pl[lo16 * 32], hi);
    #pragma unroll
    for (int j = 0; j < 8; ++j) {
      const __bf16* vr = Vt + ((size_t)kvh * HD + j * 16 + lo16) * S_LEN + kb + hi * 16;
      o[j] = wmma_bf16(pa, *(const v16bf*)vr, o[j]);
    }
  }

  #pragma unroll
  for (int r = 0; r < 8; ++r) {
    float inv = 1.0f / lrow[r];
    __bf16* orow = O + ((size_t)(q0 + hi * 8 + r) * NH + h) * HD;
    #pragma unroll
    for (int j = 0; j < 8; ++j) orow[j * 16 + lo16] = f2bf(o[j][r] * inv);
  }
}

// ---------------------------------------------------------------------------
// Host launcher
// ---------------------------------------------------------------------------
extern "C" void kernel_launch(void* const* d_in, const int* in_sizes, int n_in,
                              void* d_out, int out_size, void* d_ws, size_t ws_size,
                              hipStream_t stream) {
  const float* x    = (const float*)d_in[0];
  const float* cosb = (const float*)d_in[1];
  const float* sinb = (const float*)d_in[2];
  const float* wq   = (const float*)d_in[3];
  const float* wk   = (const float*)d_in[4];
  const float* wv   = (const float*)d_in[5];
  const float* wo   = (const float*)d_in[6];
  const float* qnw  = (const float*)d_in[7];
  const float* knw  = (const float*)d_in[8];

  char* ws = (char*)d_ws;
  const size_t SZ_XB = (size_t)S_LEN * DIM * 2;          // 16 MB
  const size_t SZ_WQ = (size_t)DIM * DIM * 2;            // 32 MB
  const size_t SZ_WK = (size_t)NKV * HD * DIM * 2;       // 8 MB
  const size_t SZ_QF = (size_t)S_LEN * NH * HD * 4;      // 32 MB
  const size_t SZ_KF = (size_t)S_LEN * NKV * HD * 4;     // 8 MB
  const size_t SZ_VB = (size_t)S_LEN * NKV * HD * 2;     // 4 MB
  const size_t SZ_QB = (size_t)S_LEN * NH * HD * 2;      // 16 MB

  size_t off = 0;
  __bf16* xb  = (__bf16*)(ws + off); off += SZ_XB;
  __bf16* wqb = (__bf16*)(ws + off); off += SZ_WQ;
  __bf16* wkb = (__bf16*)(ws + off); off += SZ_WK;
  __bf16* wvb = (__bf16*)(ws + off); off += SZ_WK;
  __bf16* wob = (__bf16*)(ws + off); off += SZ_WQ;
  float*  qf  = (float*)(ws + off);  off += SZ_QF;
  float*  kf  = (float*)(ws + off);  off += SZ_KF;
  __bf16* vb  = (__bf16*)(ws + off); off += SZ_VB;
  __bf16* qb  = (__bf16*)(ws + off); off += SZ_QB;
  __bf16* kb  = (__bf16*)(ws + off); off += SZ_VB;
  __bf16* vt  = (__bf16*)(ws + off); off += SZ_VB;
  __bf16* ao  = (__bf16*)(ws + off); off += SZ_QB;

  auto cvt = [&](const float* src, __bf16* dst, long n) {
    cvt_bf16_kernel<<<dim3((unsigned)((n + 255) / 256)), 256, 0, stream>>>(src, dst, n);
  };
  cvt(x,  xb,  (long)S_LEN * DIM);
  cvt(wq, wqb, (long)DIM * DIM);
  cvt(wk, wkb, (long)NKV * HD * DIM);
  cvt(wv, wvb, (long)NKV * HD * DIM);
  cvt(wo, wob, (long)DIM * DIM);

  // projections (TDM-staged B tiles)
  gemm_xwT_tdm<false><<<dim3(DIM / 64, S_LEN / 128), 256, 0, stream>>>(xb, wqb, qf, S_LEN, DIM, DIM);
  gemm_xwT_tdm<false><<<dim3(NKV * HD / 64, S_LEN / 128), 256, 0, stream>>>(xb, wkb, kf, S_LEN, NKV * HD, DIM);
  gemm_xwT_tdm<true ><<<dim3(NKV * HD / 64, S_LEN / 128), 256, 0, stream>>>(xb, wvb, vb, S_LEN, NKV * HD, DIM);

  // q/k rmsnorm + rope -> bf16
  rmsnorm_rope_kernel<<<dim3(S_LEN, NH), 32, 0, stream>>>(qf, cosb, sinb, qnw, qb, NH);
  rmsnorm_rope_kernel<<<dim3(S_LEN, NKV), 32, 0, stream>>>(kf, cosb, sinb, knw, kb, NKV);

  // v transpose for PV fragment loads
  {
    long n = (long)NKV * HD * S_LEN;
    transpose_v_kernel<<<dim3((unsigned)(n / 256)), 256, 0, stream>>>(vb, vt);
  }

  // flash attention
  flash_attn_kernel<<<dim3(S_LEN / 16, NH), 32, 0, stream>>>(qb, kb, vt, ao);

  // output projection -> fp32
  gemm_xwT_tdm<false><<<dim3(DIM / 64, S_LEN / 128), 256, 0, stream>>>(ao, wob, (float*)d_out, S_LEN, DIM, DIM);
}
